// TriangleMultiplication_66589172957485
// MI455X (gfx1250) — matrix operand order repaired and checked by reading, not compiled, not run on hardware
//
#include <hip/hip_runtime.h>

typedef __attribute__((ext_vector_type(2)))  unsigned int u32x2;
typedef __attribute__((ext_vector_type(4)))  unsigned int u32x4;
typedef __attribute__((ext_vector_type(8)))  unsigned int u32x8;
typedef __attribute__((ext_vector_type(8)))  float        f32x8;
typedef __attribute__((ext_vector_type(16))) __bf16       bf16x16;

#define DEV static __device__ __forceinline__

constexpr int kN = 512;
constexpr int kD = 128;
constexpr int kP = kN * kN;          // 262144 flattened positions
constexpr float kEPS = 1e-5f;

// ---------- CDNA5 async global->LDS copy (guarded; falls back to sync) ----
#if defined(__has_builtin)
#if __has_builtin(__builtin_amdgcn_global_load_async_to_lds_b128)
#define TRI_ASYNC 1
#endif
#endif
#ifndef TRI_ASYNC
#define TRI_ASYNC 0
#endif

#if defined(__has_builtin)
#if __has_builtin(__builtin_amdgcn_s_wait_asynccnt)
#define TRI_WAITA(n) __builtin_amdgcn_s_wait_asynccnt(n)
#endif
#endif
#ifndef TRI_WAITA
#if TRI_ASYNC
#define TRI_WAITA(n) asm volatile("s_wait_asynccnt %0" ::"i"(n))
#else
#define TRI_WAITA(n)
#endif
#endif

DEV void cp16(const unsigned short* g, unsigned short* l) {
#if TRI_ASYNC
  typedef int v4i __attribute__((vector_size(4 * sizeof(int))));
  __builtin_amdgcn_global_load_async_to_lds_b128((v4i*)g, (v4i*)l, 0, 0);
#else
  *(u32x4*)l = *(const u32x4*)g;
#endif
}

// ---------- bf16 helpers (RNE pack / unpack) ----------
DEV unsigned short f2bf(float f) {
  unsigned int u = __builtin_bit_cast(unsigned int, f);
  u += 0x7FFFu + ((u >> 16) & 1u);
  return (unsigned short)(u >> 16);
}
DEV unsigned int pack2(float a, float b) {
  return (unsigned int)f2bf(a) | ((unsigned int)f2bf(b) << 16);
}
DEV float bflo(unsigned int w) { return __builtin_bit_cast(float, w << 16); }
DEV float bfhi(unsigned int w) { return __builtin_bit_cast(float, w & 0xFFFF0000u); }
DEV float sig(float x) { return 1.0f / (1.0f + __expf(-x)); }

// ---------- WMMA fragment loaders from LDS ----------
// A 16x32 bf16: lane m = lane%16, half = lane/16; K runs [half*8, +8) and [16+half*8, +8).
DEV bf16x16 frag_a(const unsigned short* lds, int row0, int kofs, int rstride,
                   int lm, int half) {
  const u32x4* p = (const u32x4*)lds;
  int base = ((row0 + lm) * rstride + kofs + half * 8) >> 3;   // u32x4 units
  u32x4 lo = p[base];
  u32x4 hi = p[base + 2];
  u32x8 w = {lo.x, lo.y, lo.z, lo.w, hi.x, hi.y, hi.z, hi.w};
  return __builtin_bit_cast(bf16x16, w);
}
// B 32x16 bf16: lane n = lane%16, half = lane/16; K runs [half*8, +8) and [16+half*8, +8).
DEV bf16x16 frag_b(const unsigned short* lds, int row0, int kofs, int rstride,
                   int lm, int half) {
  const u32x4* p = (const u32x4*)lds;
  int base = ((row0 + lm) * rstride + kofs + half * 8) >> 3;
  u32x4 lo = p[base];
  u32x4 hi = p[base + 2];
  u32x8 w = {lo.x, lo.y, lo.z, lo.w, hi.x, hi.y, hi.z, hi.w};
  return __builtin_bit_cast(bf16x16, w);
}
DEV f32x8 mma(bf16x16 a, bf16x16 b, f32x8 c) {
  return __builtin_amdgcn_wmma_f32_16x16x32_bf16(false, a, false, b,
                                                 (short)0, c, false, false);
}

// =====================================================================
// Kernel A: LN(x) -> proj/gate GEMMs (K=128, Nout=640)
//   writes aT[d][p], bT[d][p], gT[d][p]  (bf16, p = i*512+k, d-major)
// =====================================================================
__global__ __launch_bounds__(256) void tri_in_kernel(
    const float* __restrict__ x,
    const float* __restrict__ nw, const float* __restrict__ nb,
    const float* __restrict__ p_in, const float* __restrict__ g_in,
    const float* __restrict__ g_out,
    unsigned short* __restrict__ aT, unsigned short* __restrict__ bT,
    unsigned short* __restrict__ gT) {
  extern __shared__ unsigned short smem[];
  unsigned short* wl = smem;                     // 640 x 128 bf16 weights
  unsigned short* nx = smem + 640 * 128;         // 128 x 128 bf16 normed rows
  float* nrm = (float*)(smem + 640 * 128 + 128 * 128);  // 128 w + 128 b
  const int tid = threadIdx.x;

  // stage 1: weights -> LDS bf16 (p_in rows 0..255, g_in 256..511, g_out 512..639)
  for (int e4 = tid; e4 < (640 * 128) / 4; e4 += 256) {
    int e = e4 * 4;
    const float* src = (e < 32768) ? (p_in + e)
                     : (e < 65536) ? (g_in + (e - 32768))
                                   : (g_out + (e - 65536));
    float4 v = *(const float4*)src;
    u32x2 pk; pk.x = pack2(v.x, v.y); pk.y = pack2(v.z, v.w);
    *(u32x2*)(wl + e) = pk;
  }
  if (tid < 128) { nrm[tid] = nw[tid]; nrm[128 + tid] = nb[tid]; }
  __syncthreads();

  // stage 2: layernorm of this block's 128 rows (2 lanes per row)
  {
    int r = tid >> 1, hf = tid & 1;
    size_t p = (size_t)blockIdx.x * 128 + r;
    const float* xr = x + p * 128 + hf * 64;
    float4 v[16];
    float s = 0.f, ss = 0.f;
#pragma unroll
    for (int e = 0; e < 16; ++e) {
      v[e] = ((const float4*)xr)[e];
      s  += v[e].x + v[e].y + v[e].z + v[e].w;
      ss += v[e].x * v[e].x + v[e].y * v[e].y + v[e].z * v[e].z + v[e].w * v[e].w;
    }
    s  += __shfl_xor(s, 1, 32);
    ss += __shfl_xor(ss, 1, 32);
    float mu = s * (1.0f / 128.0f);
    float rs = rsqrtf(ss * (1.0f / 128.0f) - mu * mu + kEPS);
#pragma unroll
    for (int e = 0; e < 16; ++e) {
      int c = hf * 64 + e * 4;
      float y0 = (v[e].x - mu) * rs * nrm[c + 0] + nrm[128 + c + 0];
      float y1 = (v[e].y - mu) * rs * nrm[c + 1] + nrm[128 + c + 1];
      float y2 = (v[e].z - mu) * rs * nrm[c + 2] + nrm[128 + c + 2];
      float y3 = (v[e].w - mu) * rs * nrm[c + 3] + nrm[128 + c + 3];
      u32x2 pk; pk.x = pack2(y0, y1); pk.y = pack2(y2, y3);
      *(u32x2*)(nx + r * 128 + c) = pk;
    }
  }
  __syncthreads();

  // stage 3: WMMA GEMMs; wave owns rows [wave*16, +16). Two ct tiles per
  // iteration -> 4 independent accumulation chains (hides WMMA RAW hazard).
  const int wave = tid >> 5, lane = tid & 31, lm = lane & 15, half = lane >> 4;
  const int r0 = wave * 16;
  const size_t p0 = (size_t)blockIdx.x * 128 + r0 + half * 8;

  bf16x16 af[4];
#pragma unroll
  for (int ks = 0; ks < 4; ++ks) af[ks] = frag_a(nx, r0, ks * 32, 128, lm, half);

  for (int cp = 0; cp < 8; ++cp) {           // ct = 2*cp + u, u in {0,1}
    f32x8 accP[2] = {}, accG[2] = {};
#pragma unroll
    for (int ks = 0; ks < 4; ++ks) {
      bf16x16 bp0 = frag_b(wl, (2 * cp + 0) * 16, ks * 32, 128, lm, half);
      bf16x16 bg0 = frag_b(wl, 256 + (2 * cp + 0) * 16, ks * 32, 128, lm, half);
      bf16x16 bp1 = frag_b(wl, (2 * cp + 1) * 16, ks * 32, 128, lm, half);
      bf16x16 bg1 = frag_b(wl, 256 + (2 * cp + 1) * 16, ks * 32, 128, lm, half);
      accP[0] = mma(af[ks], bp0, accP[0]);
      accG[0] = mma(af[ks], bg0, accG[0]);
      accP[1] = mma(af[ks], bp1, accP[1]);
      accG[1] = mma(af[ks], bg1, accG[1]);
    }
#pragma unroll
    for (int u = 0; u < 2; ++u) {
      int ct = 2 * cp + u;
      float h[8];
#pragma unroll
      for (int r = 0; r < 8; ++r) h[r] = accP[u][r] * sig(accG[u][r]);
      u32x4 st;
      st.x = pack2(h[0], h[1]); st.y = pack2(h[2], h[3]);
      st.z = pack2(h[4], h[5]); st.w = pack2(h[6], h[7]);
      unsigned short* dst = (ct < 8) ? aT : bT;
      int d = (ct & 7) * 16 + lm;
      *(u32x4*)(dst + (size_t)d * kP + p0) = st;  // 8 consecutive positions
    }
  }
  // output gate: sigmoid(x_norm @ g_out.T) -> gT[d][p]
  for (int cp = 0; cp < 4; ++cp) {
    f32x8 acc[2] = {};
#pragma unroll
    for (int ks = 0; ks < 4; ++ks) {
      bf16x16 b0 = frag_b(wl, 512 + (2 * cp + 0) * 16, ks * 32, 128, lm, half);
      bf16x16 b1 = frag_b(wl, 512 + (2 * cp + 1) * 16, ks * 32, 128, lm, half);
      acc[0] = mma(af[ks], b0, acc[0]);
      acc[1] = mma(af[ks], b1, acc[1]);
    }
#pragma unroll
    for (int u = 0; u < 2; ++u) {
      u32x4 st;
      st.x = pack2(sig(acc[u][0]), sig(acc[u][1]));
      st.y = pack2(sig(acc[u][2]), sig(acc[u][3]));
      st.z = pack2(sig(acc[u][4]), sig(acc[u][5]));
      st.w = pack2(sig(acc[u][6]), sig(acc[u][7]));
      int d = (2 * cp + u) * 16 + lm;
      *(u32x4*)(gT + (size_t)d * kP + p0) = st;
    }
  }
}

// =====================================================================
// Kernel B: per channel d: T_d = A_d x B_d^T (512^3, bf16->f32 WMMA)
//   128x128 tile / WG, K-step 32, async double-buffered LDS staging.
//   dyn LDS = 2 stages * 2 * 128*32*2 = 32768 B. grid (16, 128) x 256thr.
// =====================================================================
__global__ __launch_bounds__(256) void tri_einsum_kernel(
    const unsigned short* __restrict__ aT, const unsigned short* __restrict__ bT,
    unsigned short* __restrict__ t) {
  extern __shared__ unsigned short smem[];   // stage p at smem + p*8192
  const int tid = threadIdx.x;
  const int d = blockIdx.y;
  const int i0 = (blockIdx.x & 3) * 128;
  const int j0 = (blockIdx.x >> 2) * 128;
  const int wave = tid >> 5, lane = tid & 31, lm = lane & 15, half = lane >> 4;
  const int wi = wave & 3;   // 2 i-subtiles (32 rows)
  const int wj = wave >> 2;  // 4 j-subtiles (64 cols)
  const size_t base = (size_t)d * kP;

  auto issue_stage = [&](int ks, int p) {
    const int k0 = ks * 32;
    unsigned short* aL = smem + p * 8192;
    unsigned short* bL = aL + 4096;
#pragma unroll
    for (int it = 0; it < 2; ++it) {
      int idx = tid + it * 256;               // 512 x 16B segments per matrix
      int row = idx >> 2, seg = (idx & 3) * 8;
      cp16(aT + base + (size_t)(i0 + row) * kN + k0 + seg, aL + row * 32 + seg);
      cp16(bT + base + (size_t)(j0 + row) * kN + k0 + seg, bL + row * 32 + seg);
    }
  };

  f32x8 acc[2][4] = {};
  issue_stage(0, 0);
  for (int ks = 0; ks < 16; ++ks) {
    const int p = ks & 1;
    if (ks < 15) issue_stage(ks + 1, p ^ 1);   // prefetch next stage
    if (ks < 15) { TRI_WAITA(4); } else { TRI_WAITA(0); }
    __syncthreads();                           // stage ks resident in LDS
    const unsigned short* aL = smem + p * 8192;
    const unsigned short* bL = aL + 4096;
    bf16x16 afr[2], bfr[4];
#pragma unroll
    for (int a = 0; a < 2; ++a) afr[a] = frag_a(aL, (wi * 2 + a) * 16, 0, 32, lm, half);
#pragma unroll
    for (int b = 0; b < 4; ++b) bfr[b] = frag_b(bL, (wj * 4 + b) * 16, 0, 32, lm, half);
#pragma unroll
    for (int a = 0; a < 2; ++a)
#pragma unroll
      for (int b = 0; b < 4; ++b) acc[a][b] = mma(afr[a], bfr[b], acc[a][b]);
    __syncthreads();                           // all waves done before reuse
  }
  // epilogue: t natural layout [i][j][d], bf16
#pragma unroll
  for (int a = 0; a < 2; ++a)
#pragma unroll
    for (int b = 0; b < 4; ++b)
#pragma unroll
      for (int r = 0; r < 8; ++r) {
        int i = i0 + (wi * 2 + a) * 16 + half * 8 + r;
        int j = j0 + (wj * 4 + b) * 16 + lm;
        t[((size_t)i * kN + j) * kD + d] = f2bf(acc[a][b][r]);
      }
}

// =====================================================================
// Kernel C: LN(t) -> out-proj (128x128) -> * gate -> out (fp32)
// =====================================================================
__global__ __launch_bounds__(256) void tri_out_kernel(
    const unsigned short* __restrict__ t,
    const float* __restrict__ nw, const float* __restrict__ nb,
    const float* __restrict__ p_out,
    const unsigned short* __restrict__ gT,
    float* __restrict__ out) {
  extern __shared__ unsigned short smem[];
  unsigned short* wl = smem;                 // 128 x 128 bf16 p_out
  unsigned short* nx = smem + 128 * 128;     // 128 x 128 bf16 normed t rows
  float* nrm = (float*)(smem + 2 * 128 * 128);
  const int tid = threadIdx.x;

  for (int e4 = tid; e4 < (128 * 128) / 4; e4 += 256) {
    int e = e4 * 4;
    float4 v = *(const float4*)(p_out + e);
    u32x2 pk; pk.x = pack2(v.x, v.y); pk.y = pack2(v.z, v.w);
    *(u32x2*)(wl + e) = pk;
  }
  if (tid < 128) { nrm[tid] = nw[tid]; nrm[128 + tid] = nb[tid]; }
  __syncthreads();

  {
    int r = tid >> 1, hf = tid & 1;
    size_t p = (size_t)blockIdx.x * 128 + r;
    const unsigned short* tr = t + p * kD + hf * 64;
    float v[64];
    float s = 0.f, ss = 0.f;
#pragma unroll
    for (int e8 = 0; e8 < 8; ++e8) {
      u32x4 q = *(const u32x4*)(tr + e8 * 8);
#pragma unroll
      for (int q4 = 0; q4 < 4; ++q4) {
        unsigned int w = q[q4];
        float lo = bflo(w), hi = bfhi(w);
        v[e8 * 8 + q4 * 2 + 0] = lo;
        v[e8 * 8 + q4 * 2 + 1] = hi;
        s += lo + hi;
        ss += lo * lo + hi * hi;
      }
    }
    s  += __shfl_xor(s, 1, 32);
    ss += __shfl_xor(ss, 1, 32);
    float mu = s * (1.0f / 128.0f);
    float rs = rsqrtf(ss * (1.0f / 128.0f) - mu * mu + kEPS);
#pragma unroll
    for (int e = 0; e < 64; e += 4) {
      int c = hf * 64 + e;
      u32x2 pk;
      pk.x = pack2((v[e + 0] - mu) * rs * nrm[c + 0] + nrm[128 + c + 0],
                   (v[e + 1] - mu) * rs * nrm[c + 1] + nrm[128 + c + 1]);
      pk.y = pack2((v[e + 2] - mu) * rs * nrm[c + 2] + nrm[128 + c + 2],
                   (v[e + 3] - mu) * rs * nrm[c + 3] + nrm[128 + c + 3]);
      *(u32x2*)(nx + r * 128 + c) = pk;
    }
  }
  __syncthreads();

  const int wave = tid >> 5, lane = tid & 31, lm = lane & 15, half = lane >> 4;
  const int r0 = wave * 16;
  const size_t p0 = (size_t)blockIdx.x * 128 + r0 + half * 8;

  bf16x16 af[4];
#pragma unroll
  for (int ks = 0; ks < 4; ++ks) af[ks] = frag_a(nx, r0, ks * 32, 128, lm, half);

  for (int cp = 0; cp < 4; ++cp) {           // two ct tiles per iteration
    f32x8 acc[2] = {};
#pragma unroll
    for (int ks = 0; ks < 4; ++ks) {
      bf16x16 b0 = frag_b(wl, (2 * cp + 0) * 16, ks * 32, 128, lm, half);
      bf16x16 b1 = frag_b(wl, (2 * cp + 1) * 16, ks * 32, 128, lm, half);
      acc[0] = mma(af[ks], b0, acc[0]);
      acc[1] = mma(af[ks], b1, acc[1]);
    }
#pragma unroll
    for (int u = 0; u < 2; ++u) {
      int dch = (2 * cp + u) * 16 + lm;
      u32x4 g4 = *(const u32x4*)(gT + (size_t)dch * kP + p0);
#pragma unroll
      for (int r = 0; r < 8; ++r) {
        unsigned int w = g4[r >> 1];
        float g = (r & 1) ? bfhi(w) : bflo(w);
        out[(p0 + r) * kD + dch] = acc[u][r] * g;
      }
    }
  }
}

// =====================================================================
extern "C" void kernel_launch(void* const* d_in, const int* in_sizes, int n_in,
                              void* d_out, int out_size, void* d_ws, size_t ws_size,
                              hipStream_t stream) {
  (void)in_sizes; (void)n_in; (void)out_size; (void)ws_size;
  const float* x      = (const float*)d_in[0];
  const float* nin_w  = (const float*)d_in[1];
  const float* nin_b  = (const float*)d_in[2];
  const float* p_in   = (const float*)d_in[3];
  const float* g_in   = (const float*)d_in[4];
  const float* nout_w = (const float*)d_in[5];
  const float* nout_b = (const float*)d_in[6];
  const float* p_out  = (const float*)d_in[7];
  const float* g_out  = (const float*)d_in[8];
  float* out = (float*)d_out;

  const size_t planeBytes = (size_t)kP * kD * sizeof(unsigned short);  // 64 MiB
  unsigned short* aT = (unsigned short*)d_ws;
  unsigned short* bT = (unsigned short*)((char*)d_ws + planeBytes);
  unsigned short* gT = (unsigned short*)((char*)d_ws + 2 * planeBytes);
  unsigned short* tb = (unsigned short*)((char*)d_ws + 3 * planeBytes);

  const size_t ldsA = (640 * 128 + 128 * 128) * sizeof(unsigned short) + 256 * sizeof(float);
  const size_t ldsB = 2 * 2 * 128 * 32 * sizeof(unsigned short);   // double buffered
  const size_t ldsC = 2 * 128 * 128 * sizeof(unsigned short) + 256 * sizeof(float);

  tri_in_kernel<<<dim3(kP / 128), dim3(256), ldsA, stream>>>(
      x, nin_w, nin_b, p_in, g_in, g_out, aT, bT, gT);
  tri_einsum_kernel<<<dim3(16, kD), dim3(256), ldsB, stream>>>(aT, bT, tb);
  tri_out_kernel<<<dim3(kP / 128), dim3(256), ldsC, stream>>>(
      tb, nout_w, nout_b, p_out, gT, out);
}
